// GraphAttention_12438225289348
// MI455X (gfx1250) — matrix operand order repaired and checked by reading, not compiled, not run on hardware
//
#include <hip/hip_runtime.h>

// ---------------------------------------------------------------------------
// Criss-cross attention (N=8, C=512, H=W=96), fp32 in/out, bf16 WMMA inside.
// CDNA5 / gfx1250, wave32. All fragment traffic is 128-bit; shared operands
// are LDS-staged (async global->LDS copies, ASYNCcnt-tracked) with
// bank-conflict padding; x is streamed from HBM exactly once.
//
// Workspace layout (requires ws_size >= 227,150,336 bytes ~= 217 MiB):
//   [0,          9437184)   qT : bf16 (N, H*W, 64)   q transposed, c-contig
//   [9437184,   18874368)   kT : bf16 (N, H*W, 64)   k transposed, c-contig
//   [18874368,  94371840)   vb : bf16 (N, 512, H, W) v row-major  (w contig)
//   [94371840, 169869312)   vT : bf16 (N, 512, W, H) v col-major  (h contig)
//   [169869312,226492416)   attn: f32 (N,H,W,192) energy rows; softmax writes
//                           bf16 result into bytes [384,768) of each own row
//   [226492416,227147776)   wb : bf16 (640, 512)  fused qkv weights
//   [227147776,227150336)   bb : f32  (640)       fused qkv bias
// ---------------------------------------------------------------------------

#define NN   8
#define CCH  512
#define CQ   64
#define HH   96
#define WW   96
#define HW   (HH * WW)        // 9216
#define EDIM 192
#define NEGV (-1e30f)

#define XSTRIDE 40            // LDS stride (elems) for 32-deep K slabs
#define KSTRIDE 72            // LDS stride for 64-deep c slabs
#define ASTRIDE 104           // LDS stride for 96-deep attention slabs

#define USE_ASYNC_STAGE 1     // 1: global_load_async_to_lds_b128 staging

typedef __attribute__((ext_vector_type(16))) __bf16 v16bf;
typedef __attribute__((ext_vector_type(8)))  __bf16 bf16x8;
typedef __attribute__((ext_vector_type(8)))  float  v8f;

union FragU { v16bf v; bf16x8 h[2]; };

// float -> bf16 round-to-nearest-even
__device__ __forceinline__ __bf16 f2bf(float f) {
  union { float f; unsigned u; } in; in.f = f;
  unsigned r = in.u + 0x7FFFu + ((in.u >> 16) & 1u);
  union { unsigned short s; __bf16 b; } out;
  out.s = (unsigned short)(r >> 16);
  return out.b;
}

// 16-byte global -> LDS copy. Async path uses the CDNA5 async-DMA engine
// (no VGPR round trip, tracked by ASYNCcnt); fallback is a VGPR copy.
__device__ __forceinline__ void stage16(const void* g, void* l) {
#if USE_ASYNC_STAGE
  // VDST holds the per-lane LDS byte offset (low 32 bits of the LDS generic
  // address); VADDR is the 64-bit global address (GV mode).
  const unsigned loff = (unsigned)(unsigned long long)l;
  asm volatile("global_load_async_to_lds_b128 %0, %1, off"
               :: "v"(loff), "v"(g) : "memory");
#else
  *(bf16x8*)l = *(const bf16x8*)g;
#endif
}
__device__ __forceinline__ void stage_fence() {
#if USE_ASYNC_STAGE
  asm volatile("s_wait_asynccnt 0x0" ::: "memory");
#endif
  __syncthreads();
}

// A fragment (16x32 MxK, 16-bit): per lane, K chunks [k0+8h, +8), [k0+16+8h, +8)
__device__ __forceinline__ v16bf loadA(const __bf16* p, int k0, int half) {
  FragU f;
  f.h[0] = *(const bf16x8*)(p + k0 + 8 * half);
  f.h[1] = *(const bf16x8*)(p + k0 + 16 + 8 * half);
  return f.v;
}
// B fragment (32x16 KxN, 16-bit): per lane, K chunks [k0+8h, +8), [k0+16+8h, +8)
__device__ __forceinline__ v16bf loadB(const __bf16* p, int k0, int half) {
  FragU f;
  f.h[0] = *(const bf16x8*)(p + k0 + 8 * half);
  f.h[1] = *(const bf16x8*)(p + k0 + 16 + 8 * half);
  return f.v;
}

__device__ __forceinline__ v8f wmma_bf16(v16bf a, v16bf b, v8f c) {
  return __builtin_amdgcn_wmma_f32_16x16x32_bf16(false, a, false, b, (short)0, c,
                                                 false, false);
}

// ---------------------------------------------------------------------------
// Kernel 0: fuse + convert qkv weights to bf16 (640x512), fuse biases (640).
// ---------------------------------------------------------------------------
__global__ void __launch_bounds__(256)
prep_weights(const float* __restrict__ wq, const float* __restrict__ bq,
             const float* __restrict__ wk, const float* __restrict__ bk,
             const float* __restrict__ wv, const float* __restrict__ bv,
             __bf16* __restrict__ wb, float* __restrict__ bb) {
  const int i = blockIdx.x * 256 + threadIdx.x;     // 0 .. 640*512-1, exact
  const int r = i >> 9, c = i & 511;
  float val;
  if (r < 64)       val = wq[r * 512 + c];
  else if (r < 128) val = wk[(r - 64) * 512 + c];
  else              val = wv[(size_t)(r - 128) * 512 + c];
  wb[i] = f2bf(val);
  if (c == 0) bb[r] = (r < 64) ? bq[r] : (r < 128) ? bk[r - 64] : bv[r - 128];
}

// ---------------------------------------------------------------------------
// Kernel 1: fused QKV projection, D(640, 9216) = Wqkv x X per batch.
// One block (256 thr / 8 waves) per 16-pixel strip: x tile staged in LDS
// (f32->bf16 converted in flight, so VGPR path), x read from HBM exactly
// once; each wave computes 80 rows (5 WMMA tiles) sharing one B fragment.
// ---------------------------------------------------------------------------
__global__ void __launch_bounds__(256)
qkv_gemm(const float* __restrict__ x, const __bf16* __restrict__ wb,
         const float* __restrict__ bb,
         __bf16* __restrict__ qT, __bf16* __restrict__ kT,
         __bf16* __restrict__ vb, __bf16* __restrict__ vT) {
  __shared__ __bf16 xs[16 * XSTRIDE];               // [col][k], padded
  const int tid  = threadIdx.x;
  const int lane = tid & 31, half = lane >> 4, l16 = lane & 15;
  const int wave = tid >> 5;                        // 0..7 -> 80-row slab
  const int nt = blockIdx.x % (HW / 16);
  const int n  = blockIdx.x / (HW / 16);
  const int col0 = nt * 16;
  const int pix  = col0 + l16;

  const float* xn = x + (size_t)n * CCH * HW;
  const int scc = tid & 15, skk = tid >> 4;         // staging coords

  v8f acc[5] = {};
  const int mbase = wave * 80;

  for (int kt = 0; kt < CCH / 32; ++kt) {
    const int k0 = kt * 32;
    xs[scc * XSTRIDE + skk]      = f2bf(xn[(size_t)(k0 + skk) * HW + col0 + scc]);
    xs[scc * XSTRIDE + skk + 16] = f2bf(xn[(size_t)(k0 + skk + 16) * HW + col0 + scc]);
    if (kt < CCH / 32 - 1) {                        // hide next tile's latency
      __builtin_prefetch(&xn[(size_t)(k0 + 32 + skk) * HW + col0 + scc], 0, 0);
      __builtin_prefetch(&xn[(size_t)(k0 + 48 + skk) * HW + col0 + scc], 0, 0);
    }
    __syncthreads();
    const v16bf fb = loadB(xs + l16 * XSTRIDE, 0, half);
#pragma unroll
    for (int t = 0; t < 5; ++t) {
      const v16bf fa = loadA(wb + (size_t)(mbase + t * 16 + l16) * CCH, k0, half);
      acc[t] = wmma_bf16(fa, fb, acc[t]);
    }
    __syncthreads();
  }

#pragma unroll
  for (int t = 0; t < 5; ++t) {
    const int rbase = mbase + t * 16 + 8 * half;    // 8 consecutive rows/lane
    if (rbase < 128) {                              // q or k -> packed b128
      bf16x8 pk;
#pragma unroll
      for (int r = 0; r < 8; ++r) pk[r] = f2bf(acc[t][r] + bb[rbase + r]);
      __bf16* dst = (rbase < 64) ? qT : kT;
      const int c0 = (rbase < 64) ? rbase : rbase - 64;
      *(bf16x8*)(dst + ((size_t)n * HW + pix) * CQ + c0) = pk;
    } else {                                        // v -> both layouts
      const int h = pix / WW, w = pix % WW;
#pragma unroll
      for (int r = 0; r < 8; ++r) {
        const int c = rbase + r - 128;
        const __bf16 bv16 = f2bf(acc[t][r] + bb[rbase + r]);
        vb[((size_t)n * CCH + c) * HW + pix]         = bv16;
        vT[((size_t)n * CCH + c) * HW + w * HH + h]  = bv16;
      }
    }
  }
}

// ---------------------------------------------------------------------------
// Kernel 2: row energy. Block per (n,h): async-stage K row-slab (96x64 bf16)
// into LDS, wave mt computes E(16x96) = Q^T K; 12 WMMAs, A loaded once.
// ---------------------------------------------------------------------------
__global__ void __launch_bounds__(192)
energy_row(const __bf16* __restrict__ qT, const __bf16* __restrict__ kT,
           float* __restrict__ attn) {
  __shared__ __bf16 ks[96 * KSTRIDE];
  const int tid = threadIdx.x, lane = tid & 31, half = lane >> 4, l16 = lane & 15;
  const int wave = tid >> 5;                        // mt 0..5
  const int h = blockIdx.x % HH, n = blockIdx.x / HH;

  const __bf16* krow = kT + ((size_t)n * HW + h * WW) * CQ;   // contiguous 12KB
  for (int ch = tid; ch < 96 * 8; ch += 192) {
    const int v = ch >> 3, c8 = (ch & 7) * 8;
    stage16(krow + v * CQ + c8, ks + v * KSTRIDE + c8);
  }
  stage_fence();

  const int w = wave * 16 + l16;
  const __bf16* qrow = qT + ((size_t)n * HW + h * WW + w) * CQ;
  const v16bf fa0 = loadA(qrow, 0, half);
  const v16bf fa1 = loadA(qrow, 32, half);
  float* arow = attn + (size_t)(n * HH + h) * WW * EDIM;

#pragma unroll
  for (int ntile = 0; ntile < 6; ++ntile) {
    const __bf16* kp = ks + (ntile * 16 + l16) * KSTRIDE;
    v8f acc = {};
    acc = wmma_bf16(fa0, loadB(kp, 0, half), acc);
    acc = wmma_bf16(fa1, loadB(kp, 32, half), acc);
    const int vcol = ntile * 16 + l16;
#pragma unroll
    for (int r = 0; r < 8; ++r) {
      const int wrow = wave * 16 + r + 8 * half;
      arow[(size_t)wrow * EDIM + vcol] = acc[r];
    }
  }
}

// ---------------------------------------------------------------------------
// Kernel 3: column energy. Block per (n,w): async-stage K column-slab in LDS,
// wave mt computes E(16x96) over h; diagonal (h==j) masked to -1e30.
// ---------------------------------------------------------------------------
__global__ void __launch_bounds__(192)
energy_col(const __bf16* __restrict__ qT, const __bf16* __restrict__ kT,
           float* __restrict__ attn) {
  __shared__ __bf16 ks[96 * KSTRIDE];
  const int tid = threadIdx.x, lane = tid & 31, half = lane >> 4, l16 = lane & 15;
  const int wave = tid >> 5;
  const int w = blockIdx.x % WW, n = blockIdx.x / WW;

  for (int ch = tid; ch < 96 * 8; ch += 192) {
    const int j = ch >> 3, c8 = (ch & 7) * 8;
    stage16(kT + ((size_t)n * HW + j * WW + w) * CQ + c8, ks + j * KSTRIDE + c8);
  }
  stage_fence();

  const int hA = wave * 16 + l16;
  const __bf16* qrow = qT + ((size_t)n * HW + hA * WW + w) * CQ;
  const v16bf fa0 = loadA(qrow, 0, half);
  const v16bf fa1 = loadA(qrow, 32, half);

#pragma unroll
  for (int ntile = 0; ntile < 6; ++ntile) {
    const __bf16* kp = ks + (ntile * 16 + l16) * KSTRIDE;
    v8f acc = {};
    acc = wmma_bf16(fa0, loadB(kp, 0, half), acc);
    acc = wmma_bf16(fa1, loadB(kp, 32, half), acc);
    const int jcol = ntile * 16 + l16;
#pragma unroll
    for (int r = 0; r < 8; ++r) {
      const int hrow = wave * 16 + r + 8 * half;
      const float val = (hrow == jcol) ? NEGV : acc[r];
      attn[((size_t)(n * HH + hrow) * WW + w) * EDIM + WW + jcol] = val;
    }
  }
}

// ---------------------------------------------------------------------------
// Kernel 4: softmax over 192 logits; wave handles 2 pixels; writes bf16 result
// into bytes [384,768) of each pixel's own f32 row (wave-local, race-free).
// ---------------------------------------------------------------------------
__global__ void __launch_bounds__(256)
softmax_k(float* __restrict__ attn) {
  const int lane = threadIdx.x & 31;
  const long wq_ = (long)blockIdx.x * 8 + (threadIdx.x >> 5);  // 0..36863 exact
#pragma unroll
  for (int pp = 0; pp < 2; ++pp) {
    const long p = 2 * wq_ + pp;
    float* row = attn + p * EDIM;
    float v[6];
    float m = -3.4e38f;
#pragma unroll
    for (int i = 0; i < 6; ++i) { v[i] = row[lane + 32 * i]; m = fmaxf(m, v[i]); }
#pragma unroll
    for (int off = 16; off > 0; off >>= 1) m = fmaxf(m, __shfl_xor(m, off, 32));
    float s = 0.f;
#pragma unroll
    for (int i = 0; i < 6; ++i) { v[i] = __expf(v[i] - m); s += v[i]; }
#pragma unroll
    for (int off = 16; off > 0; off >>= 1) s += __shfl_xor(s, off, 32);
    const float inv = 1.f / s;
    __bf16* brow = (__bf16*)attn + p * 384 + 192;   // second half of own row
#pragma unroll
    for (int i = 0; i < 6; ++i) brow[lane + 32 * i] = f2bf(v[i] * inv);
  }
}

// ---------------------------------------------------------------------------
// Kernel 5: row aggregation. Block per (n,h,mb): async-stage bf16 attention
// slab (96x96) in LDS; wave computes O1(16c x 96w), 18 WMMAs; raw into d_out.
// ---------------------------------------------------------------------------
__global__ void __launch_bounds__(256)
agg_row(const __bf16* __restrict__ vb, const float* __restrict__ attn,
        float* __restrict__ out) {
  __shared__ __bf16 as[96 * ASTRIDE];
  const int tid = threadIdx.x, lane = tid & 31, half = lane >> 4, l16 = lane & 15;
  const int wave = tid >> 5;                        // 0..7
  const int mb = blockIdx.x & 3;
  const int h  = (blockIdx.x >> 2) % HH;
  const int n  = (blockIdx.x >> 2) / HH;

  const __bf16* ab = (const __bf16*)attn;
  const size_t prow = (size_t)(n * HH + h) * WW;
  for (int ch = tid; ch < 96 * 12; ch += 256) {
    const int w = ch / 12, c8 = (ch % 12) * 8;
    stage16(ab + (prow + w) * 384 + 192 + c8, as + w * ASTRIDE + c8);
  }
  stage_fence();

  const int c = mb * 128 + wave * 16 + l16;
  const __bf16* vrow = vb + ((size_t)n * CCH + c) * HW + h * WW;
  const v16bf fa0 = loadA(vrow, 0, half);
  const v16bf fa1 = loadA(vrow, 32, half);
  const v16bf fa2 = loadA(vrow, 64, half);

#pragma unroll
  for (int ntile = 0; ntile < 6; ++ntile) {
    const __bf16* ap = as + (ntile * 16 + l16) * ASTRIDE;
    v8f acc = {};
    acc = wmma_bf16(fa0, loadB(ap, 0, half), acc);
    acc = wmma_bf16(fa1, loadB(ap, 32, half), acc);
    acc = wmma_bf16(fa2, loadB(ap, 64, half), acc);
    const int wcol = ntile * 16 + l16;
#pragma unroll
    for (int r = 0; r < 8; ++r) {
      const int cr = mb * 128 + wave * 16 + r + 8 * half;
      out[((size_t)n * CCH + cr) * HW + h * WW + wcol] = acc[r];
    }
  }
}

// ---------------------------------------------------------------------------
// Kernel 6: column aggregation + finalize: out = gamma*(O1+O2) + x.
// Block per (n,w,mb): async-stage column-attention slab; A fragments from vT.
// ---------------------------------------------------------------------------
__global__ void __launch_bounds__(256)
agg_col_finalize(const __bf16* __restrict__ vT, const float* __restrict__ attn,
                 const float* __restrict__ x, const float* __restrict__ gamma,
                 float* __restrict__ out) {
  __shared__ __bf16 as[96 * ASTRIDE];
  const int tid = threadIdx.x, lane = tid & 31, half = lane >> 4, l16 = lane & 15;
  const int wave = tid >> 5;
  const int mb = blockIdx.x & 3;
  const int w  = (blockIdx.x >> 2) % WW;
  const int n  = (blockIdx.x >> 2) / WW;

  const __bf16* ab = (const __bf16*)attn;
  for (int ch = tid; ch < 96 * 12; ch += 256) {
    const int hh = ch / 12, c8 = (ch % 12) * 8;
    stage16(ab + ((size_t)(n * HH + hh) * WW + w) * 384 + 192 + 96 + c8,
            as + hh * ASTRIDE + c8);
  }
  stage_fence();

  const int c = mb * 128 + wave * 16 + l16;
  const __bf16* vrow = vT + ((size_t)n * CCH + c) * HW + w * HH;
  const v16bf fa0 = loadA(vrow, 0, half);
  const v16bf fa1 = loadA(vrow, 32, half);
  const v16bf fa2 = loadA(vrow, 64, half);
  const float g = gamma[0];

#pragma unroll
  for (int ntile = 0; ntile < 6; ++ntile) {
    const __bf16* ap = as + (ntile * 16 + l16) * ASTRIDE;
    v8f acc = {};
    acc = wmma_bf16(fa0, loadB(ap, 0, half), acc);
    acc = wmma_bf16(fa1, loadB(ap, 32, half), acc);
    acc = wmma_bf16(fa2, loadB(ap, 64, half), acc);
    const int hcol = ntile * 16 + l16;
#pragma unroll
    for (int r = 0; r < 8; ++r) {
      const int cr = mb * 128 + wave * 16 + r + 8 * half;
      const size_t idx = ((size_t)n * CCH + cr) * HW + (size_t)hcol * WW + w;
      out[idx] = g * (out[idx] + acc[r]) + x[idx];
    }
  }
}

// ---------------------------------------------------------------------------
extern "C" void kernel_launch(void* const* d_in, const int* in_sizes, int n_in,
                              void* d_out, int out_size, void* d_ws, size_t ws_size,
                              hipStream_t stream) {
  const float* x     = (const float*)d_in[0];
  const float* wq    = (const float*)d_in[1];
  const float* bq    = (const float*)d_in[2];
  const float* wk    = (const float*)d_in[3];
  const float* bk    = (const float*)d_in[4];
  const float* wv    = (const float*)d_in[5];
  const float* bv    = (const float*)d_in[6];
  const float* gamma = (const float*)d_in[7];
  float* out = (float*)d_out;

  char* ws = (char*)d_ws;
  __bf16* qT   = (__bf16*)(ws);
  __bf16* kT   = (__bf16*)(ws + 9437184);
  __bf16* vb   = (__bf16*)(ws + 18874368);
  __bf16* vT   = (__bf16*)(ws + 94371840);
  float*  attn = (float*)(ws + 169869312);
  __bf16* wb   = (__bf16*)(ws + 226492416);
  float*  bb   = (float*)(ws + 227147776);

  prep_weights<<<1280, 256, 0, stream>>>(wq, bq, wk, bk, wv, bv, wb, bb);
  qkv_gemm<<<NN * (HW / 16), 256, 0, stream>>>(x, wb, bb, qT, kT, vb, vT);
  energy_row<<<NN * HH, 192, 0, stream>>>(qT, kT, attn);
  energy_col<<<NN * WW, 192, 0, stream>>>(qT, kT, attn);
  softmax_k<<<4608, 256, 0, stream>>>(attn);
  agg_row<<<NN * HH * 4, 256, 0, stream>>>(vb, attn, out);
  agg_col_finalize<<<NN * WW * 4, 256, 0, stream>>>(vT, attn, x, gamma, out);
}
